// Mixture_of_Models_41326175322733
// MI455X (gfx1250) — matrix operand order repaired and checked by reading, not compiled, not run on hardware
//
#include <hip/hip_runtime.h>
#include <hip/hip_bf16.h>
#include <math.h>

// Problem constants (from reference)
static const int N_  = 4;
static const int S_  = 1024;
static const int K_  = 4;
static const int V_  = 32000;
static const int C_  = 1000;
static const int D_  = 512;
static const int NH_ = 8;
static const int NB_ = 6;
static const int HD_ = 2048;
static const int DH_ = 64;   // D/NH
static const int CAP_ = 256; // S/K

typedef __attribute__((ext_vector_type(16))) _Float16 v16h;
typedef __attribute__((ext_vector_type(8)))  float    v8f;

// ---------------------------------------------------------------------------
// Generic batched GEMM: C = epi(alpha * A@B + bias [+ R]), f16 WMMA, f32 acc.
// Tile: BM=128, BN=128, BK=32. 256 threads = 8 waves (wave32), wave grid 4x2,
// each wave computes a 32x64 sub-tile = 2x4 v_wmma_f32_16x16x32_f16 accums.
// transB: B[d,n] = Bp[n*ldb + d]  (used for Q @ K^T)
// epi: 0=none, 1=+bias, 2=+bias,gelu(tanh), 3=+bias,+R residual
// ---------------------------------------------------------------------------
__global__ __launch_bounds__(256)
void gemm_wmma_kernel(const float* __restrict__ A, long long lda, long long strideA,
                      const float* __restrict__ B, long long ldb, long long strideB, int transB,
                      const float* __restrict__ bias,
                      const float* __restrict__ R, long long ldr, long long strideR,
                      float* __restrict__ C, long long ldc, long long strideC,
                      int M, int Nc, int Kd, float alpha, int epi)
{
    __shared__ _Float16 As[128][40];   // [m][k], K-contiguous, padded
    __shared__ _Float16 Bs[128][40];   // [n][k], K-contiguous, padded

    const int tid  = threadIdx.x;
    const int lane = tid & 31;
    const int wave = tid >> 5;
    const int wm   = wave >> 1;              // 0..3 : 32-row band
    const int wn   = wave & 1;               // 0..1 : 64-col band
    const int bm   = blockIdx.x * 128;
    const int bn   = blockIdx.y * 128;
    const long long bz = blockIdx.z;

    A += bz * strideA;
    B += bz * strideB;
    C += bz * strideC;
    if (epi == 3) R += bz * strideR;

    v8f acc[2][4];
    #pragma unroll
    for (int r = 0; r < 2; ++r)
        #pragma unroll
        for (int c = 0; c < 4; ++c)
            #pragma unroll
            for (int i = 0; i < 8; ++i) acc[r][c][i] = 0.0f;

    for (int k0 = 0; k0 < Kd; k0 += 32) {
        // ---- stage A tile 128x32 (f32 -> f16) ----
        #pragma unroll
        for (int u = 0; u < 4; ++u) {
            int q  = tid * 4 + u;            // 1024 float4 loads
            int ra = q >> 3;                 // 0..127
            int ca = (q & 7) << 2;           // 0..28
            float4 f = *(const float4*)(A + (long long)(bm + ra) * lda + (k0 + ca));
            As[ra][ca + 0] = (_Float16)f.x;
            As[ra][ca + 1] = (_Float16)f.y;
            As[ra][ca + 2] = (_Float16)f.z;
            As[ra][ca + 3] = (_Float16)f.w;
        }
        if (k0 + 32 < Kd)
            __builtin_prefetch(A + (long long)(bm + (tid >> 3)) * lda + (k0 + 32), 0, 1);

        // ---- stage B tile into Bs[n][k] ----
        if (!transB) {
            #pragma unroll
            for (int u = 0; u < 4; ++u) {
                int q  = tid * 4 + u;
                int kr = q >> 5;             // 0..31
                int cn = (q & 31) << 2;      // 0..124
                int col = bn + cn;
                float4 f;
                const float* bp = B + (long long)(k0 + kr) * ldb + col;
                if (col + 3 < Nc) {
                    f = *(const float4*)bp;
                } else {
                    f.x = (col + 0 < Nc) ? bp[0] : 0.0f;
                    f.y = (col + 1 < Nc) ? bp[1] : 0.0f;
                    f.z = (col + 2 < Nc) ? bp[2] : 0.0f;
                    f.w = (col + 3 < Nc) ? bp[3] : 0.0f;
                }
                Bs[cn + 0][kr] = (_Float16)f.x;
                Bs[cn + 1][kr] = (_Float16)f.y;
                Bs[cn + 2][kr] = (_Float16)f.z;
                Bs[cn + 3][kr] = (_Float16)f.w;
            }
        } else {
            #pragma unroll
            for (int u = 0; u < 4; ++u) {
                int q  = tid * 4 + u;
                int nr = q >> 3;             // 0..127
                int ck = (q & 7) << 2;       // 0..28
                int col = bn + nr;
                float4 f = make_float4(0.f, 0.f, 0.f, 0.f);
                if (col < Nc)
                    f = *(const float4*)(B + (long long)col * ldb + (k0 + ck));
                Bs[nr][ck + 0] = (_Float16)f.x;
                Bs[nr][ck + 1] = (_Float16)f.y;
                Bs[nr][ck + 2] = (_Float16)f.z;
                Bs[nr][ck + 3] = (_Float16)f.w;
            }
        }
        __syncthreads();

        // ---- build fragments (contiguous k: lane half h holds k 16h..16h+15 in A and B) and issue WMMAs ----
        v16h afrag[2], bfrag[4];
        const int kbA = (lane < 16) ? 0 : 16;  // A: low lanes K0-15, high K16-31
        #pragma unroll
        for (int r = 0; r < 2; ++r) {
            const _Float16* ap = &As[wm * 32 + r * 16 + (lane & 15)][kbA];
            #pragma unroll
            for (int i = 0; i < 16; ++i) afrag[r][i] = ap[i];
        }
        const int kbB = (lane < 16) ? 0 : 16;  // B: low lanes K0-15, high K16-31
        #pragma unroll
        for (int c = 0; c < 4; ++c) {
            const _Float16* bp = &Bs[wn * 64 + c * 16 + (lane & 15)][kbB];
            #pragma unroll
            for (int i = 0; i < 16; ++i) bfrag[c][i] = bp[i];
        }
        #pragma unroll
        for (int r = 0; r < 2; ++r)
            #pragma unroll
            for (int c = 0; c < 4; ++c)
                acc[r][c] = __builtin_amdgcn_wmma_f32_16x16x32_f16(
                    false, afrag[r], false, bfrag[c],
                    (short)0, acc[r][c], false, false);

        __syncthreads();
    }

    // ---- epilogue: C/D layout -> lane = N, VGPR i = M(+8 for high lanes) ----
    #pragma unroll
    for (int r = 0; r < 2; ++r) {
        #pragma unroll
        for (int c = 0; c < 4; ++c) {
            int row0 = bm + wm * 32 + r * 16 + ((lane < 16) ? 0 : 8);
            int col  = bn + wn * 64 + c * 16 + (lane & 15);
            if (col < Nc) {
                float bv = (epi >= 1) ? bias[col] : 0.0f;
                #pragma unroll
                for (int i = 0; i < 8; ++i) {
                    float v = alpha * acc[r][c][i] + bv;
                    if (epi == 2) {
                        float t = v + 0.044715f * v * v * v;
                        v = 0.5f * v * (1.0f + tanhf(0.7978845608028654f * t));
                    }
                    long long row = row0 + i;
                    if (epi == 3) v += R[row * ldr + col];
                    C[row * ldc + col] = v;
                }
            }
        }
    }
}

// ---------------------------------------------------------------------------
// x[k,n,s,:] = tok_emb[k, x_in[n,s], :] + pos_emb[k, s, :]
// ---------------------------------------------------------------------------
__global__ __launch_bounds__(256)
void embed_kernel(const int* __restrict__ x_in, const float* __restrict__ tok_emb,
                  const float* __restrict__ pos_emb, float* __restrict__ x)
{
    int g = blockIdx.x;                 // k*N*S + n*S + s
    int k = g / (N_ * S_);
    int r = g - k * N_ * S_;
    int n = r / S_;
    int s = r - n * S_;
    int tok = x_in[n * S_ + s];
    const float* te = tok_emb + ((long long)k * V_ + tok) * D_;
    const float* pe = pos_emb + ((long long)k * S_ + s) * D_;
    float* out = x + ((long long)k * N_ * S_ + r) * D_;
    for (int d = threadIdx.x; d < D_; d += 256) out[d] = te[d] + pe[d];
}

// ---------------------------------------------------------------------------
// LayerNorm over last dim (ddof=0, eps 1e-5): y = (x-mean)*rsqrt(var+eps)*s+b
// ---------------------------------------------------------------------------
__global__ __launch_bounds__(256)
void layernorm_kernel(const float* __restrict__ x, const float* __restrict__ sc,
                      const float* __restrict__ bi, float* __restrict__ y, int D)
{
    __shared__ float red[256];
    long long row = blockIdx.x;
    const float* xr = x + row * D;
    float s = 0.f;
    for (int d = threadIdx.x; d < D; d += 256) s += xr[d];
    red[threadIdx.x] = s; __syncthreads();
    for (int off = 128; off > 0; off >>= 1) {
        if (threadIdx.x < off) red[threadIdx.x] += red[threadIdx.x + off];
        __syncthreads();
    }
    float mean = red[0] / D;
    __syncthreads();
    float v = 0.f;
    for (int d = threadIdx.x; d < D; d += 256) { float t = xr[d] - mean; v += t * t; }
    red[threadIdx.x] = v; __syncthreads();
    for (int off = 128; off > 0; off >>= 1) {
        if (threadIdx.x < off) red[threadIdx.x] += red[threadIdx.x + off];
        __syncthreads();
    }
    float rstd = rsqrtf(red[0] / D + 1e-5f);
    float* yr = y + row * D;
    for (int d = threadIdx.x; d < D; d += 256)
        yr[d] = (xr[d] - mean) * rstd * sc[d] + bi[d];
}

// ---------------------------------------------------------------------------
// In-place causal softmax on att rows: row = (b)*S + q over [N*NH*S] rows.
// ---------------------------------------------------------------------------
__global__ __launch_bounds__(256)
void causal_softmax_kernel(float* __restrict__ att)
{
    __shared__ float red[256];
    long long row = blockIdx.x;
    int q = (int)(row % S_);
    float* a = att + row * S_;
    float vals[4];
    float m = -1e30f;
    #pragma unroll
    for (int u = 0; u < 4; ++u) {
        int j = threadIdx.x + u * 256;
        float v = (j <= q) ? a[j] : -1e30f;
        vals[u] = v;
        m = fmaxf(m, v);
    }
    red[threadIdx.x] = m; __syncthreads();
    for (int off = 128; off > 0; off >>= 1) {
        if (threadIdx.x < off) red[threadIdx.x] = fmaxf(red[threadIdx.x], red[threadIdx.x + off]);
        __syncthreads();
    }
    m = red[0]; __syncthreads();
    float sum = 0.f;
    #pragma unroll
    for (int u = 0; u < 4; ++u) {
        float e = __expf(vals[u] - m);   // masked -> exp(-huge) = 0
        vals[u] = e;
        sum += e;
    }
    red[threadIdx.x] = sum; __syncthreads();
    for (int off = 128; off > 0; off >>= 1) {
        if (threadIdx.x < off) red[threadIdx.x] += red[threadIdx.x + off];
        __syncthreads();
    }
    float inv = 1.f / red[0];
    #pragma unroll
    for (int u = 0; u < 4; ++u) a[threadIdx.x + u * 256] = vals[u] * inv;
}

// ---------------------------------------------------------------------------
// Per-token CE on logits: loss[n,k,s] = logsumexp(clf[k,n,s,:]) - clf[...,y]
// ---------------------------------------------------------------------------
__global__ __launch_bounds__(256)
void ce_loss_kernel(const float* __restrict__ clf, const int* __restrict__ y,
                    float* __restrict__ loss)
{
    __shared__ float red[256];
    int g = blockIdx.x;                 // k*N*S + n*S + s
    int k = g / (N_ * S_);
    int r = g - k * N_ * S_;
    int n = r / S_;
    int s = r - n * S_;
    const float* xr = clf + (long long)g * C_;
    float m = -1e30f;
    for (int j = threadIdx.x; j < C_; j += 256) m = fmaxf(m, xr[j]);
    red[threadIdx.x] = m; __syncthreads();
    for (int off = 128; off > 0; off >>= 1) {
        if (threadIdx.x < off) red[threadIdx.x] = fmaxf(red[threadIdx.x], red[threadIdx.x + off]);
        __syncthreads();
    }
    m = red[0]; __syncthreads();
    float sum = 0.f;
    for (int j = threadIdx.x; j < C_; j += 256) sum += __expf(xr[j] - m);
    red[threadIdx.x] = sum; __syncthreads();
    for (int off = 128; off > 0; off >>= 1) {
        if (threadIdx.x < off) red[threadIdx.x] += red[threadIdx.x + off];
        __syncthreads();
    }
    if (threadIdx.x == 0) {
        int t = y[n * S_ + s];
        loss[((long long)n * K_ + k) * S_ + s] = logf(red[0]) + m - xr[t];
    }
}

// ---------------------------------------------------------------------------
// Confidence head: conf[n,k,s] = dot(xln[n,s,:], Wconf[k]) + bconf[k]
// ---------------------------------------------------------------------------
__global__ __launch_bounds__(128)
void conf_kernel(const float* __restrict__ xln, const float* __restrict__ w,
                 const float* __restrict__ b, float* __restrict__ conf_out, int k)
{
    __shared__ float red[128];
    int g = blockIdx.x;                 // n*S + s
    const float* xr = xln + (long long)g * D_;
    float s = 0.f;
    for (int d = threadIdx.x; d < D_; d += 128) s += xr[d] * w[d];
    red[threadIdx.x] = s; __syncthreads();
    for (int off = 64; off > 0; off >>= 1) {
        if (threadIdx.x < off) red[threadIdx.x] += red[threadIdx.x + off];
        __syncthreads();
    }
    if (threadIdx.x == 0) {
        int n = g / S_;
        int ss = g - n * S_;
        conf_out[((long long)n * K_ + k) * S_ + ss] = red[0] + b[0];
    }
}

// ---------------------------------------------------------------------------
// Capacity-constrained auction routing (serial per batch; stable argsort K=4)
// ---------------------------------------------------------------------------
__global__ void routing_kernel(const float* __restrict__ loss, int* __restrict__ chosen,
                               float* __restrict__ idx_out)
{
    int n = threadIdx.x;
    if (n >= N_) return;
    int cnt[K_] = {0, 0, 0, 0};
    for (int s = 0; s < S_; ++s) {
        float l[K_];
        for (int k = 0; k < K_; ++k) l[k] = loss[((long long)n * K_ + k) * S_ + s];
        int ord[K_] = {0, 1, 2, 3};
        for (int i = 1; i < K_; ++i) {          // stable insertion sort ascending
            int oi = ord[i]; float key = l[oi]; int j = i - 1;
            while (j >= 0 && l[ord[j]] > key) { ord[j + 1] = ord[j]; --j; }
            ord[j + 1] = oi;
        }
        int pick = ord[0];
        for (int j = 0; j < K_; ++j)
            if (cnt[ord[j]] < CAP_) { pick = ord[j]; break; }
        cnt[pick]++;
        chosen[n * S_ + s]  = pick;
        idx_out[n * S_ + s] = (float)pick;
    }
}

// ---------------------------------------------------------------------------
// out[n,s,c] = clf[chosen[n,s], n, s, c]
// ---------------------------------------------------------------------------
__global__ __launch_bounds__(256)
void gather_kernel(const float* __restrict__ clf, const int* __restrict__ chosen,
                   float* __restrict__ out)
{
    long long i = (long long)blockIdx.x * 256 + threadIdx.x;  // over N*S*C
    if (i >= (long long)N_ * S_ * C_) return;
    long long row = i / C_;
    int c = (int)(i - row * C_);
    int k = chosen[row];
    out[i] = clf[((long long)k * N_ * S_ + row) * C_ + c];
}

// ---------------------------------------------------------------------------
extern "C" void kernel_launch(void* const* d_in, const int* in_sizes, int n_in,
                              void* d_out, int out_size, void* d_ws, size_t ws_size,
                              hipStream_t stream)
{
    (void)in_sizes; (void)n_in; (void)out_size; (void)ws_size;

    const int*   x_in    = (const int*)  d_in[0];
    const int*   y_t     = (const int*)  d_in[1];
    const float* tok_emb = (const float*)d_in[2];
    const float* pos_emb = (const float*)d_in[3];
    const float* ln1_s   = (const float*)d_in[4];
    const float* ln1_b   = (const float*)d_in[5];
    const float* Wqkv    = (const float*)d_in[6];
    const float* bqkv    = (const float*)d_in[7];
    const float* Wo      = (const float*)d_in[8];
    const float* bo      = (const float*)d_in[9];
    const float* ln2_s   = (const float*)d_in[10];
    const float* ln2_b   = (const float*)d_in[11];
    const float* W1      = (const float*)d_in[12];
    const float* b1      = (const float*)d_in[13];
    const float* W2      = (const float*)d_in[14];
    const float* b2      = (const float*)d_in[15];
    const float* lnf_s   = (const float*)d_in[16];
    const float* lnf_b   = (const float*)d_in[17];
    const float* Wcls    = (const float*)d_in[18];
    const float* bcls    = (const float*)d_in[19];
    const float* Wconf   = (const float*)d_in[20];
    const float* bconf   = (const float*)d_in[21];

    const long long NS = (long long)N_ * S_;  // 4096

    // ---- workspace carve (≈309 MB) ----
    char* wp = (char*)d_ws;
    auto carve = [&](size_t bytes) -> void* {
        void* p = (void*)wp;
        wp += (bytes + 255) & ~(size_t)255;
        return p;
    };
    float* x    = (float*)carve((size_t)K_ * NS * D_ * 4);        // residual streams
    float* hln  = (float*)carve((size_t)NS * D_ * 4);             // LN output / xln
    float* qkv  = (float*)carve((size_t)NS * 3 * D_ * 4);
    float* obuf = (float*)carve((size_t)NS * D_ * 4);
    float* mid  = (float*)carve((size_t)NS * HD_ * 4);
    float* att  = (float*)carve((size_t)N_ * NH_ * S_ * S_ * 4);
    float* clf  = (float*)carve((size_t)K_ * NS * C_ * 4);
    float* loss = (float*)carve((size_t)N_ * K_ * S_ * 4);
    int*   chosen = (int*)carve((size_t)N_ * S_ * 4);

    float* out_logits = (float*)d_out;                 // [N,S,C]
    float* out_conf   = out_logits + NS * C_;          // [N,K,S]
    float* out_idx    = out_conf + (long long)N_ * K_ * S_;  // [N,S]

    auto gemm = [&](const float* A, long long lda, long long sA,
                    const float* B, long long ldb, long long sB, int tB,
                    const float* bias,
                    const float* R, long long ldr, long long sR,
                    float* Cc, long long ldc, long long sC,
                    int M, int Nc, int Kd, float alpha, int epi, int batch) {
        dim3 grid(M / 128, (Nc + 127) / 128, batch);
        hipLaunchKernelGGL(gemm_wmma_kernel, grid, dim3(256), 0, stream,
                           A, lda, sA, B, ldb, sB, tB, bias, R, ldr, sR,
                           Cc, ldc, sC, M, Nc, Kd, alpha, epi);
    };

    hipLaunchKernelGGL(embed_kernel, dim3(K_ * N_ * S_), dim3(256), 0, stream,
                       x_in, tok_emb, pos_emb, x);

    for (int k = 0; k < K_; ++k) {
        float* xk = x + (long long)k * NS * D_;
        for (int blk = 0; blk < NB_; ++blk) {
            long long wb = (long long)k * NB_ + blk;

            hipLaunchKernelGGL(layernorm_kernel, dim3((unsigned)NS), dim3(256), 0, stream,
                               xk, ln1_s + wb * D_, ln1_b + wb * D_, hln, D_);

            // qkv = hln @ Wqkv + bqkv
            gemm(hln, D_, 0,
                 Wqkv + wb * (long long)D_ * 3 * D_, 3 * D_, 0, 0,
                 bqkv + wb * 3 * D_, nullptr, 0, 0,
                 qkv, 3 * D_, 0, (int)NS, 3 * D_, D_, 1.0f, 1, 1);

            // scores = Q @ K^T / sqrt(DH), batched over heads, per batch n
            for (int n = 0; n < N_; ++n) {
                const float* qb = qkv + (long long)n * S_ * 3 * D_;
                gemm(qb, 3 * D_, DH_,
                     qb + D_, 3 * D_, DH_, 1,
                     nullptr, nullptr, 0, 0,
                     att + (long long)n * NH_ * S_ * S_, S_, (long long)S_ * S_,
                     S_, S_, DH_, 0.125f, 0, NH_);
            }
            hipLaunchKernelGGL(causal_softmax_kernel, dim3(N_ * NH_ * S_), dim3(256), 0, stream, att);

            // o = P @ V, batched over heads
            for (int n = 0; n < N_; ++n) {
                const float* vb = qkv + (long long)n * S_ * 3 * D_ + 2 * D_;
                gemm(att + (long long)n * NH_ * S_ * S_, S_, (long long)S_ * S_,
                     vb, 3 * D_, DH_, 0,
                     nullptr, nullptr, 0, 0,
                     obuf + (long long)n * S_ * D_, D_, DH_,
                     S_, DH_, S_, 1.0f, 0, NH_);
            }

            // x = x + o @ Wo + bo   (in-place residual)
            gemm(obuf, D_, 0,
                 Wo + wb * (long long)D_ * D_, D_, 0, 0,
                 bo + wb * D_, xk, D_, 0,
                 xk, D_, 0, (int)NS, D_, D_, 1.0f, 3, 1);

            hipLaunchKernelGGL(layernorm_kernel, dim3((unsigned)NS), dim3(256), 0, stream,
                               xk, ln2_s + wb * D_, ln2_b + wb * D_, hln, D_);

            // mid = gelu(hln @ W1 + b1)
            gemm(hln, D_, 0,
                 W1 + wb * (long long)D_ * HD_, HD_, 0, 0,
                 b1 + wb * HD_, nullptr, 0, 0,
                 mid, HD_, 0, (int)NS, HD_, D_, 1.0f, 2, 1);

            // x = x + mid @ W2 + b2
            gemm(mid, HD_, 0,
                 W2 + wb * (long long)HD_ * D_, D_, 0, 0,
                 b2 + wb * D_, xk, D_, 0,
                 xk, D_, 0, (int)NS, D_, HD_, 1.0f, 3, 1);
        }

        // final LN
        hipLaunchKernelGGL(layernorm_kernel, dim3((unsigned)NS), dim3(256), 0, stream,
                           xk, lnf_s + (long long)k * D_, lnf_b + (long long)k * D_, hln, D_);

        // classifier logits for model k
        gemm(hln, D_, 0,
             Wcls + (long long)k * D_ * C_, C_, 0, 0,
             bcls + (long long)k * C_, nullptr, 0, 0,
             clf + (long long)k * NS * C_, C_, 0, (int)NS, C_, D_, 1.0f, 1, 1);

        // confidence head
        hipLaunchKernelGGL(conf_kernel, dim3((unsigned)NS), dim3(128), 0, stream,
                           hln, Wconf + (long long)k * D_, bconf + k, out_conf, k);
    }

    // routing on detached logits
    hipLaunchKernelGGL(ce_loss_kernel, dim3(K_ * N_ * S_), dim3(256), 0, stream, clf, y_t, loss);
    hipLaunchKernelGGL(routing_kernel, dim3(1), dim3(32), 0, stream, loss, chosen, out_idx);
    hipLaunchKernelGGL(gather_kernel,
                       dim3((unsigned)(((long long)N_ * S_ * C_ + 255) / 256)), dim3(256), 0, stream,
                       clf, chosen, out_logits);
}
